// OuterProductMean_51170240365056
// MI455X (gfx1250) — matrix operand order repaired and checked by reading, not compiled, not run on hardware
//
#include <hip/hip_runtime.h>
#include <math.h>

// ---------------- problem constants ----------------
#define S_DIM 128
#define N_RES 256
#define D_MSA 256
#define D_HID 32
#define D_PAIR 128

typedef __attribute__((ext_vector_type(16))) __bf16 v16bf;
typedef __attribute__((ext_vector_type(8)))  __bf16 v8bf;
typedef __attribute__((ext_vector_type(8)))  float  v8f;

// =====================================================================
// Kernel 1: layernorm + dual 256->32 projection, scattered into WMMA
// operand layouts (bf16).  NOTE the role swap vs. the naive version:
// the main kernel computes outer^T[e,c] = sum_s b[j,s,e] * a[i,s,c]
// so that accumulator r-indices run along contiguous k = c*32+e.
//   a_g : B-layout [i(256)][chalf(2)][kstep(4)][lane(32)][v(16)]
//         element a[i,s,c]: c = 16*chalf + lane%16, s = 32*kstep + 8*(lane/16) + v%8 + 16*(v/8)
//   b_g : A-layout [j(256)][ehalf(2)][kstep(4)][lane(32)][v(16)]
//         element b[j,s,e]: e = 16*ehalf + lane%16,
//         s = 32*kstep + Kmap(lane/16, v)  (A 16x32 half-lane K interleave)
// =====================================================================
__global__ __launch_bounds__(128) void k_prep_rows(
    const float* __restrict__ m, const float* __restrict__ mask,
    const float* __restrict__ ln_w, const float* __restrict__ ln_b,
    const float* __restrict__ w1, const float* __restrict__ b1,
    const float* __restrict__ w2, const float* __restrict__ b2,
    __bf16* __restrict__ a_g, __bf16* __restrict__ b_g)
{
    __shared__ float xrow[4][D_MSA];
    const int lane = threadIdx.x & 31;
    const int wv   = threadIdx.x >> 5;
    const int row  = blockIdx.x * 4 + wv;      // row = s*256 + n
    const int s    = row >> 8;
    const int n    = row & 255;

    const float* xp = m + (size_t)row * D_MSA;
    float xv[8], s1 = 0.f, s2 = 0.f;
#pragma unroll
    for (int q = 0; q < 8; q++) {
        float v = xp[lane + 32 * q];
        xv[q] = v; s1 += v; s2 += v * v;
    }
#pragma unroll
    for (int off = 16; off > 0; off >>= 1) {
        s1 += __shfl_xor(s1, off, 32);
        s2 += __shfl_xor(s2, off, 32);
    }
    const float mu   = s1 * (1.f / D_MSA);
    const float var  = s2 * (1.f / D_MSA) - mu * mu;
    const float rstd = rsqrtf(var + 1e-5f);
#pragma unroll
    for (int q = 0; q < 8; q++) {
        int d = lane + 32 * q;
        xrow[wv][d] = (xv[q] - mu) * rstd * ln_w[d] + ln_b[d];
    }
    // intra-wave LDS RAW: DS ops are in-order within a wave.

    const int h = lane;                        // output channel 0..31
    float acc1 = 0.f, acc2 = 0.f;
    const float* w1r = w1 + h * D_MSA;
    const float* w2r = w2 + h * D_MSA;
    for (int d = 0; d < D_MSA; d++) {
        float x = xrow[wv][d];
        acc1 = fmaf(x, w1r[d], acc1);
        acc2 = fmaf(x, w2r[d], acc2);
    }
    const float mk = mask[row] * 0.0625f;      // N^-0.5 = 1/16
    const float av = (acc1 + b1[h]) * mk;
    const float bv = (acc2 + b2[h]) * mk;

    const int Kp = s & 31, kstep = s >> 5;
    {   // a -> B-layout (N = c)
        int ch = h >> 4, ncol = h & 15;
        int g  = (Kp >> 3) & 1, v = (Kp & 7) | ((Kp >> 1) & 8);
        int ln = g * 16 + ncol;
        size_t idx = ((((size_t)n * 2 + ch) * 4 + kstep) * 32 + ln) * 16 + v;
        a_g[idx] = (__bf16)av;
    }
    {   // b -> A-layout (M = e)
        int eh = h >> 4, mrow = h & 15;
        int g  = (Kp >> 3) & 1;
        int v  = (Kp & 7) | ((Kp >> 1) & 8);
        int ln = g * 16 + mrow;
        size_t idx = ((((size_t)n * 2 + eh) * 4 + kstep) * 32 + ln) * 16 + v;
        b_g[idx] = (__bf16)bv;
    }
}

// =====================================================================
// Kernel 2a: w_out [128 x 1024] -> bf16 B-layout [ktile(32)][ptile(8)][lane][v]
// =====================================================================
__global__ __launch_bounds__(256) void k_prep_wout(
    const float* __restrict__ w_out, __bf16* __restrict__ wout_B)
{
    int idx = blockIdx.x * 256 + threadIdx.x;  // 131072 elems
    int p = idx >> 10, k = idx & 1023;
    int pt = p >> 4, lm = p & 15;
    int kt = k >> 5, r = k & 31, g = (r >> 3) & 1, v = (r & 7) | ((r >> 1) & 8);
    int ln = g * 16 + lm;
    wout_B[(((size_t)kt * 8 + pt) * 32 + ln) * 16 + v] = (__bf16)w_out[idx];
}

// =====================================================================
// Kernel 2b: wz [128 x 128] -> bf16 B-layout [kstep(4)][qtile(8)][lane][v]
// =====================================================================
__global__ __launch_bounds__(256) void k_prep_wz(
    const float* __restrict__ wz, __bf16* __restrict__ wz_B)
{
    int idx = blockIdx.x * 256 + threadIdx.x;  // 16384 elems
    int q = idx >> 7, p = idx & 127;
    int ks = p >> 5, r = p & 31, g = (r >> 3) & 1, v = (r & 7) | ((r >> 1) & 8);
    int qt = q >> 4;
    int ln = g * 16 + (q & 15);
    wz_B[(((size_t)ks * 8 + qt) * 32 + ln) * 16 + v] = (__bf16)wz[idx];
}

// =====================================================================
// Kernel 2c: norm[i,j] = sum_s mk[s,i]*mk[s,j]
// =====================================================================
__global__ __launch_bounds__(256) void k_prep_norm(
    const float* __restrict__ mask, float* __restrict__ normmat)
{
    int i = blockIdx.x, j = threadIdx.x;
    float s = 0.f;
    for (int ss = 0; ss < S_DIM; ss++)
        s += mask[ss * N_RES + i] * mask[ss * N_RES + j];
    normmat[i * N_RES + j] = s * (0.0625f * 0.0625f);
}

// =====================================================================
// Main fused kernel: one wave per (i, 32 consecutive j).
//   Phase 1: 32 pairs x (2x2 tiles x 4 ksteps) bf16 WMMA -> outer^T,
//            packed v8bf ds_store_b128 into LDS stage [t(32)][k(1024)]
//   Phase 2: projection GEMM 32x128x1024 as two M-tiles sharing each
//            w_out B tile (halves L2 traffic per pair)
//   Phase 3: +b_out, /(eps+norm), exact GELU, LN(128) -> bf16 LDS
//   Phase 4: final GEMM 32x128x128 with wz (shared B tiles), +bz, store
// LDS: 64 KB stage + 16 KB zbuf = 80 KB -> 4 single-wave WGs per WGP.
// =====================================================================
__global__ __launch_bounds__(32) void k_main(
    const __bf16* __restrict__ a_g, const __bf16* __restrict__ b_g,
    const __bf16* __restrict__ wout_B, const __bf16* __restrict__ wz_B,
    const float* __restrict__ normmat, const float* __restrict__ b_out,
    const float* __restrict__ ln_ow, const float* __restrict__ ln_ob,
    const float* __restrict__ bz, float* __restrict__ out)
{
    extern __shared__ char smem[];
    __bf16* stage = (__bf16*)smem;              // 32*1024 bf16 = 64 KB
    float*  zbuf  = (float*)(smem + 65536);     // 32*128 f32  = 16 KB

    const int lane = threadIdx.x;               // single wave
    const int gw   = blockIdx.x;                // 2048 wave-groups
    const int i    = gw >> 3;
    const int j0   = (gw & 7) << 5;             // 32 j per wave
    const int grp  = lane >> 4;                 // lane half
    const int lm   = lane & 15;

    // ---- a (B-operand) for row-block i resident: 8 x v16bf = 64 VGPRs ----
    v16bf aop[2][4];
#pragma unroll
    for (int ch = 0; ch < 2; ch++)
#pragma unroll
        for (int ks = 0; ks < 4; ks++)
            aop[ch][ks] = *(const v16bf*)(a_g + ((((size_t)i * 2 + ch) * 4 + ks) * 32 + lane) * 16);

    // ---- Phase 1: per-pair 32x32 outer^T via WMMA, packed stage to LDS ----
    for (int t = 0; t < 32; t++) {
        const int j = j0 + t;
        v8f acc[2][2] = {};                     // [eh][ch], D = outer^T tile
#pragma unroll
        for (int ks = 0; ks < 4; ks++) {
            v16bf bop0 = *(const v16bf*)(b_g + ((((size_t)j * 2 + 0) * 4 + ks) * 32 + lane) * 16);
            v16bf bop1 = *(const v16bf*)(b_g + ((((size_t)j * 2 + 1) * 4 + ks) * 32 + lane) * 16);
#pragma unroll
            for (int ch = 0; ch < 2; ch++) {
                acc[0][ch] = __builtin_amdgcn_wmma_f32_16x16x32_bf16(
                    false, bop0, false, aop[ch][ks], (short)0, acc[0][ch], false, false);
                acc[1][ch] = __builtin_amdgcn_wmma_f32_16x16x32_bf16(
                    false, bop1, false, aop[ch][ks], (short)0, acc[1][ch], false, false);
            }
        }
        // D element r of lane L: e = 16*eh + 8*(L/16) + r,  c = 16*ch + L%16
        // -> k = c*32 + e contiguous along r: one packed 16B store per tile
#pragma unroll
        for (int eh = 0; eh < 2; eh++)
#pragma unroll
            for (int ch = 0; ch < 2; ch++) {
                v8bf pk;
#pragma unroll
                for (int r = 0; r < 8; r++) pk[r] = (__bf16)acc[eh][ch][r];
                int k0 = (16 * ch + lm) * 32 + 16 * eh + 8 * grp;
                *(v8bf*)(stage + t * 1024 + k0) = pk;
            }
    }

    // ---- Phase 2: Z[32 x 128] = stage[32 x 1024] * w_out^T ----
    // two M-tiles (pairs 0-15 / 16-31) share every B tile
    for (int pt = 0; pt < 8; pt++) {
        v8f p0 = {}, p1 = {};
        for (int kt = 0; kt < 32; kt++) {
            v16bf A0, A1;
            ((v8bf*)&A0)[0] = *(const v8bf*)(stage + lm * 1024 + kt * 32 + grp * 8);
            ((v8bf*)&A0)[1] = *(const v8bf*)(stage + lm * 1024 + kt * 32 + 16 + grp * 8);
            ((v8bf*)&A1)[0] = *(const v8bf*)(stage + (16 + lm) * 1024 + kt * 32 + grp * 8);
            ((v8bf*)&A1)[1] = *(const v8bf*)(stage + (16 + lm) * 1024 + kt * 32 + 16 + grp * 8);
            v16bf B = *(const v16bf*)(wout_B + (((size_t)kt * 8 + pt) * 32 + lane) * 16);
            p0 = __builtin_amdgcn_wmma_f32_16x16x32_bf16(
                false, A0, false, B, (short)0, p0, false, false);
            p1 = __builtin_amdgcn_wmma_f32_16x16x32_bf16(
                false, A1, false, B, (short)0, p1, false, false);
        }
        const float bo = b_out[pt * 16 + lm];
#pragma unroll
        for (int r = 0; r < 8; r++) {
            int tr = grp * 8 + r;
            float nv0 = normmat[i * N_RES + j0 + tr];
            float nv1 = normmat[i * N_RES + j0 + 16 + tr];
            float z0 = (p0[r] + bo) / (1e-3f + nv0);
            float z1 = (p1[r] + bo) / (1e-3f + nv1);
            z0 = 0.5f * z0 * (1.f + erff(z0 * 0.70710678118654752f));  // exact GELU
            z1 = 0.5f * z1 * (1.f + erff(z1 * 0.70710678118654752f));
            zbuf[tr * 128 + pt * 16 + lm]        = z0;
            zbuf[(16 + tr) * 128 + pt * 16 + lm] = z1;
        }
    }

    // ---- Phase 3: LayerNorm over 128 per pair, bf16 back into LDS ----
    __bf16* stage2 = stage;   // reuse first 8 KB (phase 2 fully consumed it)
    for (int t = 0; t < 32; t++) {
        float vals[4], s1 = 0.f, s2 = 0.f;
#pragma unroll
        for (int q = 0; q < 4; q++) {
            float v = zbuf[t * 128 + lane + 32 * q];
            vals[q] = v; s1 += v; s2 += v * v;
        }
#pragma unroll
        for (int off = 16; off > 0; off >>= 1) {
            s1 += __shfl_xor(s1, off, 32);
            s2 += __shfl_xor(s2, off, 32);
        }
        float mu   = s1 * (1.f / D_PAIR);
        float rstd = rsqrtf(s2 * (1.f / D_PAIR) - mu * mu + 1e-5f);
#pragma unroll
        for (int q = 0; q < 4; q++) {
            int p = lane + 32 * q;
            stage2[t * 128 + p] = (__bf16)((vals[q] - mu) * rstd * ln_ow[p] + ln_ob[p]);
        }
    }

    // ---- Phase 4: O[32 x 128] = zn * wz^T, +bz, f32 store ----
    for (int qt = 0; qt < 8; qt++) {
        v8f f0 = {}, f1 = {};
#pragma unroll
        for (int ks = 0; ks < 4; ks++) {
            v16bf A0, A1;
            ((v8bf*)&A0)[0] = *(const v8bf*)(stage2 + lm * 128 + ks * 32 + grp * 8);
            ((v8bf*)&A0)[1] = *(const v8bf*)(stage2 + lm * 128 + ks * 32 + 16 + grp * 8);
            ((v8bf*)&A1)[0] = *(const v8bf*)(stage2 + (16 + lm) * 128 + ks * 32 + grp * 8);
            ((v8bf*)&A1)[1] = *(const v8bf*)(stage2 + (16 + lm) * 128 + ks * 32 + 16 + grp * 8);
            v16bf B = *(const v16bf*)(wz_B + (((size_t)ks * 8 + qt) * 32 + lane) * 16);
            f0 = __builtin_amdgcn_wmma_f32_16x16x32_bf16(
                false, A0, false, B, (short)0, f0, false, false);
            f1 = __builtin_amdgcn_wmma_f32_16x16x32_bf16(
                false, A1, false, B, (short)0, f1, false, false);
        }
        const float bzv = bz[qt * 16 + lm];
#pragma unroll
        for (int r = 0; r < 8; r++) {
            int jr0 = j0 + grp * 8 + r;
            int jr1 = j0 + 16 + grp * 8 + r;
            out[((size_t)i * N_RES + jr0) * D_PAIR + qt * 16 + lm] = f0[r] + bzv;
            out[((size_t)i * N_RES + jr1) * D_PAIR + qt * 16 + lm] = f1[r] + bzv;
        }
    }
}

// =====================================================================
// Launch
// =====================================================================
extern "C" void kernel_launch(void* const* d_in, const int* in_sizes, int n_in,
                              void* d_out, int out_size, void* d_ws, size_t ws_size,
                              hipStream_t stream)
{
    (void)in_sizes; (void)n_in; (void)out_size; (void)ws_size;
    const float* m     = (const float*)d_in[0];
    const float* mask  = (const float*)d_in[1];
    const float* ln_w  = (const float*)d_in[2];
    const float* ln_b  = (const float*)d_in[3];
    const float* w1    = (const float*)d_in[4];
    const float* b1    = (const float*)d_in[5];
    const float* w2    = (const float*)d_in[6];
    const float* b2    = (const float*)d_in[7];
    const float* w_out = (const float*)d_in[8];
    const float* b_out = (const float*)d_in[9];
    const float* ln_ow = (const float*)d_in[10];
    const float* ln_ob = (const float*)d_in[11];
    const float* wz    = (const float*)d_in[12];
    const float* bz    = (const float*)d_in[13];

    char* ws = (char*)d_ws;
    __bf16* a_g    = (__bf16*)(ws);                              // 2 MB
    __bf16* b_g    = (__bf16*)(ws + (size_t)(2u << 20));         // 2 MB
    __bf16* wout_B = (__bf16*)(ws + (size_t)(4u << 20));         // 256 KB
    __bf16* wz_B   = (__bf16*)(ws + (size_t)(4u << 20) + 262144);// 32 KB
    float*  normmat= (float*) (ws + (size_t)(4u << 20) + 262144 + 32768); // 256 KB

    k_prep_rows<<<8192, 128, 0, stream>>>(m, mask, ln_w, ln_b, w1, b1, w2, b2, a_g, b_g);
    k_prep_wout<<<512, 256, 0, stream>>>(w_out, wout_B);
    k_prep_wz  <<<64,  256, 0, stream>>>(wz, wz_B);
    k_prep_norm<<<256, 256, 0, stream>>>(mask, normmat);
    k_main<<<2048, 32, 81920, stream>>>(a_g, b_g, wout_B, wz_B, normmat,
                                        b_out, ln_ow, ln_ob, bz, (float*)d_out);
}
